// LocalitySelfAttention_18313740550732
// MI455X (gfx1250) — matrix operand order repaired and checked by reading, not compiled, not run on hardware
//
#include <hip/hip_runtime.h>
#include <hip/hip_bf16.h>

// ---------------------------------------------------------------------------
// LocalitySelfAttention on gfx1250 (MI455X), wave32 + WMMA f16 (f32 accum).
//   B=2, N=4096 (64x64 grid), DIM=256, H=8, hd=32, window=7 (+-3).
// Pipeline:
//   1) fp32 -> f16 conversion of x, W_qkv, W_proj
//   2) QKV GEMM  (v_wmma_f32_16x16x32_f16) with DOUBLE-BUFFERED
//      global_load_async_to_lds_b128 tile staging (ASYNCcnt-tracked DMA),
//      scatter to q[b,h,n,d], k[b,h,n,d], v TRANSPOSED vT[b,h,d,n]
//   3) fused local attention, one wave32 per (b,h,16-query block), flash-style
//      online softmax; S = Q*K^T via one WMMA (K dim == hd == 32);
//      O accumulated as O^T = V^T * P^T via WMMA (P routed through LDS)
//   4) proj GEMM (WMMA, async-staged) + bias, fp32 output
// ---------------------------------------------------------------------------

typedef _Float16 h8  __attribute__((ext_vector_type(8)));
typedef _Float16 h16 __attribute__((ext_vector_type(16)));
typedef float    f8  __attribute__((ext_vector_type(8)));

#define WMMA_F16(a, b, c) \
  __builtin_amdgcn_wmma_f32_16x16x32_f16(false, (a), false, (b), (short)0, (c), false, false)

static __device__ __forceinline__ h16 cat8(h8 lo, h8 hi) {
  return __builtin_shufflevector(lo, hi, 0,1,2,3,4,5,6,7,8,9,10,11,12,13,14,15);
}

// Async 16B global -> LDS copy (per-lane), tracked by ASYNCcnt.
// Low 32 bits of a generic pointer to a __shared__ object are the DS address.
static __device__ __forceinline__ void async_b128(void* lds, const void* gptr) {
  unsigned lds_addr = (unsigned)(unsigned long long)lds;
  asm volatile("global_load_async_to_lds_b128 %0, %1, off"
               :: "v"(lds_addr), "v"(gptr)
               : "memory");
}
static __device__ __forceinline__ void wait_async0() {
  asm volatile("s_wait_asynccnt 0x0" ::: "memory");
}

// ------------------------------- conversion --------------------------------
__global__ void cvt_f32_f16(const float* __restrict__ in, _Float16* __restrict__ out, int n) {
  int i = blockIdx.x * blockDim.x + threadIdx.x;
  if (i < n) out[i] = (_Float16)in[i];
}

// ------------------------------- QKV GEMM ----------------------------------
// C[8192 x 768] = A[8192 x 256] * W^T (W row-major [768 x 256], NT GEMM).
// Block tile 128(M) x 64(N), 8 waves as 4x2, each wave 32x32 (2x2 WMMA tiles).
// Double-buffered async LDS staging.
__global__ __launch_bounds__(256) void gemm_qkv(
    const _Float16* __restrict__ A, const _Float16* __restrict__ W,
    _Float16* __restrict__ qh, _Float16* __restrict__ kh, _Float16* __restrict__ vT) {
  __shared__ __align__(16) _Float16 As[2][128][40];
  __shared__ __align__(16) _Float16 Bs[2][64][40];
  const int tid   = threadIdx.x;
  const int wave  = tid >> 5;
  const int lane  = tid & 31;
  const int l16   = lane & 15;
  const bool laneLo = (lane < 16);
  const int wm = wave >> 1;          // 0..3
  const int wn = wave & 1;           // 0..1
  const int blockM = blockIdx.y * 128;
  const int blockN = blockIdx.x * 64;

  const int arow0 = (tid      ) >> 2, aseg0 = (tid      ) & 3;
  const int arow1 = (tid + 256) >> 2, aseg1 = (tid + 256) & 3;
  const int brow  = tid >> 2,         bseg  = tid & 3;

#define STAGE_TILE(buf, kk)                                                           \
  do {                                                                                \
    async_b128(&As[buf][arow0][aseg0 * 8], &A[(blockM + arow0) * 256 + (kk) + aseg0 * 8]); \
    async_b128(&As[buf][arow1][aseg1 * 8], &A[(blockM + arow1) * 256 + (kk) + aseg1 * 8]); \
    async_b128(&Bs[buf][brow][bseg * 8],   &W[(blockN + brow) * 256 + (kk) + bseg * 8]);   \
  } while (0)

  STAGE_TILE(0, 0);
  wait_async0();
  __syncthreads();

  f8 acc[2][2] = {};
  for (int step = 0; step < 8; ++step) {
    const int cur = step & 1;
    if (step < 7) STAGE_TILE(cur ^ 1, (step + 1) * 32);

    h16 af[2], bf[2];
    const int ak = laneLo ? 0 : 8;              // A layout K offsets
    const int bk = laneLo ? 0 : 8;              // B layout K offsets
#pragma unroll
    for (int i = 0; i < 2; ++i) {
      int row = wm * 32 + i * 16 + l16;
      af[i] = cat8(*(const h8*)&As[cur][row][ak], *(const h8*)&As[cur][row][ak + 16]);
    }
#pragma unroll
    for (int j = 0; j < 2; ++j) {
      int row = wn * 32 + j * 16 + l16;
      bf[j] = cat8(*(const h8*)&Bs[cur][row][bk], *(const h8*)&Bs[cur][row][bk + 16]);
    }
#pragma unroll
    for (int i = 0; i < 2; ++i)
#pragma unroll
      for (int j = 0; j < 2; ++j)
        acc[i][j] = WMMA_F16(af[i], bf[j], acc[i][j]);

    wait_async0();       // this wave's async loads for next buffer landed
    __syncthreads();     // all waves done reading cur / staging next
  }
#undef STAGE_TILE

  // scatter-store: col c -> (q|k|v, head, dim); v stored transposed [b,h,d,n]
#pragma unroll
  for (int i = 0; i < 2; ++i)
#pragma unroll
    for (int j = 0; j < 2; ++j)
#pragma unroll
      for (int r = 0; r < 8; ++r) {
        int m = blockM + wm * 32 + i * 16 + r + (laneLo ? 0 : 8);
        int c = blockN + wn * 32 + j * 16 + l16;
        int b = m >> 12, n = m & 4095;
        int three = c >> 8, rem = c & 255, hh = rem >> 5, d = rem & 31;
        _Float16 hv = (_Float16)acc[i][j][r];
        if (three == 0)      qh[((b * 8 + hh) * 4096 + n) * 32 + d] = hv;
        else if (three == 1) kh[((b * 8 + hh) * 4096 + n) * 32 + d] = hv;
        else                 vT[((b * 8 + hh) * 32 + d) * 4096 + n] = hv;
      }
}

// ----------------------------- local attention -----------------------------
// One wave32 per (b, h, 16-query block). 16 consecutive tokens share a grid
// row; neighborhood = rows qr-3..qr+3, cols [c_lo, c_hi] (<=22), covered by
// two overlapping 16-key chunks (chunk1 deduped via col >= c_lo+16 mask).
__global__ __launch_bounds__(32) void attn_local(
    const _Float16* __restrict__ qh, const _Float16* __restrict__ kh,
    const _Float16* __restrict__ vT, const float* __restrict__ temp,
    _Float16* __restrict__ attn_h) {
  __shared__ __align__(16) _Float16 Pbuf[16][32];
  __shared__ float sf_s[16];
  __shared__ float l_s[16];

  const int bid  = blockIdx.x;
  const int qblk = bid & 255;
  const int h    = (bid >> 8) & 7;
  const int b    = bid >> 11;
  const int lane = threadIdx.x;
  const int l16  = lane & 15;
  const bool laneLo = (lane < 16);

  const int q0  = qblk * 16;
  const int qr  = q0 >> 6;
  const int qc0 = q0 & 63;
  const float alpha = 0.17677669529663689f * temp[0];   // hd^-0.5 * temperature

  const _Float16* qbase = qh + (size_t)(b * 8 + h) * 4096 * 32;
  const _Float16* kbase = kh + (size_t)(b * 8 + h) * 4096 * 32;
  const _Float16* vbase = vT + (size_t)(b * 8 + h) * 32 * 4096;

  // Q fragment (A layout): lane row = q0+l16, two contiguous 16B loads
  h16 qf;
  {
    const _Float16* p = qbase + (q0 + l16) * 32 + (laneLo ? 0 : 8);
    qf = cat8(*(const h8*)p, *(const h8*)(p + 16));
  }

  float mrow[8], lrow[8];
#pragma unroll
  for (int r = 0; r < 8; ++r) { mrow[r] = -__builtin_inff(); lrow[r] = 0.0f; }
  f8 accO[2] = {};

  const int c_lo = (qc0 - 3 < 0) ? 0 : (qc0 - 3);
  const int c_hi = (qc0 + 18 > 63) ? 63 : (qc0 + 18);

  for (int kr = qr - 3; kr <= qr + 3; ++kr) {
    if (kr < 0 || kr > 63) continue;
    const int t0 = kr * 64 + c_lo;          // chunk0 first token
    const int t1 = kr * 64 + (c_hi - 15);   // chunk1 first token (overlaps, deduped)

    // ---- scores: S_c = Q * K_c^T (one WMMA each, K dim = hd = 32) ----
    f8 s[2];
#pragma unroll
    for (int cch = 0; cch < 2; ++cch) {
      int tok = (cch ? t1 : t0) + l16;      // B layout: N (key) = lane%16
      const _Float16* p = kbase + tok * 32 + (laneLo ? 0 : 8);
      h16 kf = cat8(*(const h8*)p, *(const h8*)(p + 16));
      f8 z = {};
      s[cch] = WMMA_F16(qf, kf, z);
    }

    // ---- mask + scale (C layout: q = r + half*8, key col = lane%16) ----
#pragma unroll
    for (int cch = 0; cch < 2; ++cch) {
      int col = (cch ? (c_hi - 15) : c_lo) + l16;
#pragma unroll
      for (int r = 0; r < 8; ++r) {
        int qi = r + (laneLo ? 0 : 8);
        int dc = col - (qc0 + qi);
        bool valid = (dc >= -3) && (dc <= 3) && (cch == 0 || col >= c_lo + 16);
        s[cch][r] = valid ? s[cch][r] * alpha : -__builtin_inff();
      }
    }

    // ---- online softmax, write P (f16) to LDS ----
#pragma unroll
    for (int r = 0; r < 8; ++r) {
      float rm = fmaxf(s[0][r], s[1][r]);
#pragma unroll
      for (int msk = 1; msk < 16; msk <<= 1) rm = fmaxf(rm, __shfl_xor(rm, msk, 32));
      float mo = mrow[r];
      float mn = fmaxf(mo, rm);
      float sc = __expf(mo - mn);           // exp(-inf - finite) == 0 first pass
      float p0 = __expf(s[0][r] - mn);
      float p1 = __expf(s[1][r] - mn);
      int qi = r + (laneLo ? 0 : 8);
      Pbuf[qi][l16]      = (_Float16)p0;
      Pbuf[qi][16 + l16] = (_Float16)p1;
      float ps = p0 + p1;
#pragma unroll
      for (int msk = 1; msk < 16; msk <<= 1) ps += __shfl_xor(ps, msk, 32);
      lrow[r] = lrow[r] * sc + ps;
      mrow[r] = mn;
      if (l16 == 0) sf_s[qi] = sc;
    }
    __syncthreads();

    // ---- rescale O^T accumulators (column = query = lane%16) ----
    float f = sf_s[l16];
#pragma unroll
    for (int r = 0; r < 8; ++r) { accO[0][r] *= f; accO[1][r] *= f; }

    // P^T B-fragment: lane reads Pbuf[query=l16][keys 0..7,16..23 | 8..15,24..31]
    h16 pf;
    {
      const _Float16* pp = &Pbuf[l16][laneLo ? 0 : 8];
      pf = cat8(*(const h8*)pp, *(const h8*)(pp + 16));
    }

    // ---- O^T += V^T * P^T  (two WMMAs: dims 0..15 and 16..31) ----
#pragma unroll
    for (int db = 0; db < 2; ++db) {
      int d = db * 16 + l16;                // A layout: M (dim) = lane%16
      const _Float16* vrow = vbase + d * 4096;
      h8 lo, hi;                            // tokens may be odd -> 2B-safe loads
      __builtin_memcpy(&lo, vrow + t0 + (laneLo ? 0 : 8), 16);
      __builtin_memcpy(&hi, vrow + t1 + (laneLo ? 0 : 8), 16);
      h16 vf = cat8(lo, hi);
      accO[db] = WMMA_F16(vf, pf, accO[db]);
    }
    __syncthreads();
  }

  // ---- finalize: divide by l, store [b, token, h*32 + d] as f16 ----
  if (l16 == 0) {
#pragma unroll
    for (int r = 0; r < 8; ++r) l_s[r + (laneLo ? 0 : 8)] = lrow[r];
  }
  __syncthreads();
  float inv = 1.0f / l_s[l16];
  int tok = q0 + l16;
  _Float16* outp = attn_h + (size_t)(b * 4096 + tok) * 256 + h * 32;
#pragma unroll
  for (int db = 0; db < 2; ++db)
#pragma unroll
    for (int r = 0; r < 8; ++r) {
      int d = db * 16 + r + (laneLo ? 0 : 8);
      outp[d] = (_Float16)(accO[db][r] * inv);
    }
}

// ------------------------------- proj GEMM ---------------------------------
// out[8192 x 256] = attn[8192 x 256] * Wproj^T + bias   (fp32 output)
__global__ __launch_bounds__(256) void gemm_proj(
    const _Float16* __restrict__ A, const _Float16* __restrict__ W,
    const float* __restrict__ bias, float* __restrict__ out) {
  __shared__ __align__(16) _Float16 As[2][128][40];
  __shared__ __align__(16) _Float16 Bs[2][64][40];
  const int tid  = threadIdx.x;
  const int wave = tid >> 5;
  const int lane = tid & 31;
  const int l16  = lane & 15;
  const bool laneLo = (lane < 16);
  const int wm = wave >> 1;
  const int wn = wave & 1;
  const int blockM = blockIdx.y * 128;
  const int blockN = blockIdx.x * 64;

  const int arow0 = (tid      ) >> 2, aseg0 = (tid      ) & 3;
  const int arow1 = (tid + 256) >> 2, aseg1 = (tid + 256) & 3;
  const int brow  = tid >> 2,         bseg  = tid & 3;

#define STAGE_TILE(buf, kk)                                                           \
  do {                                                                                \
    async_b128(&As[buf][arow0][aseg0 * 8], &A[(blockM + arow0) * 256 + (kk) + aseg0 * 8]); \
    async_b128(&As[buf][arow1][aseg1 * 8], &A[(blockM + arow1) * 256 + (kk) + aseg1 * 8]); \
    async_b128(&Bs[buf][brow][bseg * 8],   &W[(blockN + brow) * 256 + (kk) + bseg * 8]);   \
  } while (0)

  STAGE_TILE(0, 0);
  wait_async0();
  __syncthreads();

  f8 acc[2][2] = {};
  for (int step = 0; step < 8; ++step) {
    const int cur = step & 1;
    if (step < 7) STAGE_TILE(cur ^ 1, (step + 1) * 32);

    h16 af[2], bf[2];
    const int ak = laneLo ? 0 : 8;
    const int bk = laneLo ? 0 : 8;
#pragma unroll
    for (int i = 0; i < 2; ++i) {
      int row = wm * 32 + i * 16 + l16;
      af[i] = cat8(*(const h8*)&As[cur][row][ak], *(const h8*)&As[cur][row][ak + 16]);
    }
#pragma unroll
    for (int j = 0; j < 2; ++j) {
      int row = wn * 32 + j * 16 + l16;
      bf[j] = cat8(*(const h8*)&Bs[cur][row][bk], *(const h8*)&Bs[cur][row][bk + 16]);
    }
#pragma unroll
    for (int i = 0; i < 2; ++i)
#pragma unroll
      for (int j = 0; j < 2; ++j)
        acc[i][j] = WMMA_F16(af[i], bf[j], acc[i][j]);

    wait_async0();
    __syncthreads();
  }
#undef STAGE_TILE

#pragma unroll
  for (int i = 0; i < 2; ++i)
#pragma unroll
    for (int j = 0; j < 2; ++j)
#pragma unroll
      for (int r = 0; r < 8; ++r) {
        int m = blockM + wm * 32 + i * 16 + r + (laneLo ? 0 : 8);
        int c = blockN + wn * 32 + j * 16 + l16;
        out[(size_t)m * 256 + c] = acc[i][j][r] + bias[c];
      }
}

// ------------------------------- launcher ----------------------------------
extern "C" void kernel_launch(void* const* d_in, const int* in_sizes, int n_in,
                              void* d_out, int out_size, void* d_ws, size_t ws_size,
                              hipStream_t stream) {
  const float* x     = (const float*)d_in[0];   // [2,4096,256]
  const float* Wqkv  = (const float*)d_in[1];   // [768,256]
  const float* Wproj = (const float*)d_in[2];   // [256,256]
  const float* bproj = (const float*)d_in[3];   // [256]
  const float* temp  = (const float*)d_in[4];   // [1]
  float* out = (float*)d_out;                   // [2,4096,256] fp32

  char* ws = (char*)d_ws;
  const size_t SZ_X   = (size_t)8192 * 256 * 2;         // 4 MB
  const size_t SZ_WQ  = (size_t)768 * 256 * 2;
  const size_t SZ_WP  = (size_t)256 * 256 * 2;
  const size_t SZ_QKV = (size_t)2 * 8 * 4096 * 32 * 2;  // 4 MB each
  _Float16* x_h     = (_Float16*)(ws);
  _Float16* wqkv_h  = (_Float16*)(ws + SZ_X);
  _Float16* wproj_h = (_Float16*)(ws + SZ_X + SZ_WQ);
  _Float16* q_h     = (_Float16*)(ws + SZ_X + SZ_WQ + SZ_WP);
  _Float16* k_h     = (_Float16*)(ws + SZ_X + SZ_WQ + SZ_WP + SZ_QKV);
  _Float16* vT_h    = (_Float16*)(ws + SZ_X + SZ_WQ + SZ_WP + 2 * SZ_QKV);
  _Float16* attn_h  = (_Float16*)(ws + SZ_X + SZ_WQ + SZ_WP + 3 * SZ_QKV);

  cvt_f32_f16<<<(2097152 + 255) / 256, 256, 0, stream>>>(x, x_h, 2097152);
  cvt_f32_f16<<<(196608 + 255) / 256, 256, 0, stream>>>(Wqkv, wqkv_h, 196608);
  cvt_f32_f16<<<(65536 + 255) / 256, 256, 0, stream>>>(Wproj, wproj_h, 65536);

  gemm_qkv<<<dim3(12, 64), 256, 0, stream>>>(x_h, wqkv_h, q_h, k_h, vT_h);

  attn_local<<<4096, 32, 0, stream>>>(q_h, k_h, vT_h, temp, attn_h);

  gemm_proj<<<dim3(4, 64), 256, 0, stream>>>(attn_h, wproj_h, bproj, out);
}
